// GNNTime_11622181503214
// MI455X (gfx1250) — compile-verified, matrix operand order checked by reading, not run on hardware
//
#include <hip/hip_runtime.h>
#include <hip/hip_bf16.h>

typedef float v2f  __attribute__((ext_vector_type(2)));
typedef float v8f  __attribute__((ext_vector_type(8)));
typedef _Float16 v16h __attribute__((ext_vector_type(16)));

#define FDIM 64   // DIN == DH == DOUT == 64

// ---------------------------------------------------------------------------
// Degree / normalization kernels
// ---------------------------------------------------------------------------
__global__ void gcn_deg_init(float* __restrict__ deg, int n) {
    int i = blockIdx.x * blockDim.x + threadIdx.x;
    if (i < n) deg[i] = 1.0f;                 // self-loop weight
}

__global__ void gcn_deg_acc(const int* __restrict__ col,
                            const float* __restrict__ w,
                            float* __restrict__ deg, int e) {
    int i = blockIdx.x * blockDim.x + threadIdx.x;
    if (i < e) atomicAdd(&deg[col[i]], w[i]);
}

__global__ void gcn_dinv(const float* __restrict__ deg,
                         float* __restrict__ dinv, int n) {
    int i = blockIdx.x * blockDim.x + threadIdx.x;
    if (i < n) {
        float d = deg[i];                     // d >= 1 always (self-loop)
        dinv[i] = __frsqrt_rn(d);
    }
}

// ---------------------------------------------------------------------------
// WMMA GEMM: H[n x 64] = X[n x 64] @ W^T   (W is [64 x 64] row-major, W[j][k])
// One wave computes one 16x16 tile of H. 8 waves/block -> 32 rows x 64 cols.
// EXEC must be all-1s for WMMA: loads are clamped, only stores predicated.
// ---------------------------------------------------------------------------
__global__ __launch_bounds__(256)
void gcn_gemm64(const float* __restrict__ X, const float* __restrict__ W,
                float* __restrict__ H, int n) {
    const int lane = threadIdx.x & 31;
    const int wave = threadIdx.x >> 5;
    const int rowBase = blockIdx.x * 32 + (wave >> 2) * 16;  // 2 row tiles
    const int colBase = (wave & 3) * 16;                     // 4 col tiles
    const int half = lane >> 4;       // 0: K pair {0,1} | 1: K pair {2,3}
    const int l15  = lane & 15;

    int arow = rowBase + l15;
    if (arow >= n) arow = n - 1;      // clamp (store is guarded instead)
    const float* __restrict__ xrow = X + (size_t)arow * FDIM;
    const float* __restrict__ wrow = W + (size_t)(colBase + l15) * FDIM;

    v8f c = {0.f, 0.f, 0.f, 0.f, 0.f, 0.f, 0.f, 0.f};

#if __has_builtin(__builtin_amdgcn_wmma_f32_16x16x4_f32)
    // Full-precision fp32 WMMA path: V_WMMA_F32_16X16X4_F32, K step 4.
    #pragma unroll
    for (int k0 = 0; k0 < FDIM; k0 += 4) {
        const int ka = k0 + half * 2;
        v2f a, b;
        a.x = xrow[ka]; a.y = xrow[ka + 1];      // A: 16x4, row-striped per lane
        b.x = wrow[ka]; b.y = wrow[ka + 1];      // B[k][j] = W[j][k]
        c = __builtin_amdgcn_wmma_f32_16x16x4_f32(
                false, a, false, b, (short)0, c, false, false);
    }
#else
    // Fallback: f16 WMMA (codegen-confirmed builtin), K step 32.
    #pragma unroll
    for (int k0 = 0; k0 < FDIM; k0 += 32) {
        v16h a, b;
        #pragma unroll
        for (int i = 0; i < 16; ++i) {
            const int p = i >> 1, wlo = i & 1;
            const int k = k0 + (p >> 2) * 16 + half * 8 + (p & 3) * 2 + wlo;
            a[i] = (_Float16)xrow[k];
            b[i] = (_Float16)wrow[k];
        }
        c = __builtin_amdgcn_wmma_f32_16x16x32_f16(
                false, a, false, b, (short)0, c, false, false);
    }
#endif

    // C/D layout: VGPR r -> rows r (lanes 0-15) and r+8 (lanes 16-31), N = l15
    #pragma unroll
    for (int r = 0; r < 8; ++r) {
        const int m = rowBase + half * 8 + r;
        if (m < n) H[(size_t)m * FDIM + colBase + l15] = c[r];
    }
}

// ---------------------------------------------------------------------------
// agg[i] = dinv[i]^2 * hlin[i] + bias   (self-loop term + bias pre-added)
// float4-vectorized: n*16 threads.
// ---------------------------------------------------------------------------
__global__ void gcn_init_agg(const float* __restrict__ H,
                             const float* __restrict__ dinv,
                             const float* __restrict__ bias,
                             float* __restrict__ out, int n) {
    int t = blockIdx.x * blockDim.x + threadIdx.x;
    if (t >= n * 16) return;
    const int i = t >> 4, c4 = t & 15;
    const float d2 = dinv[i] * dinv[i];
    float4 h = reinterpret_cast<const float4*>(H)[t];
    float4 b = reinterpret_cast<const float4*>(bias)[c4];
    float4 o;
    o.x = d2 * h.x + b.x;  o.y = d2 * h.y + b.y;
    o.z = d2 * h.z + b.z;  o.w = d2 * h.w + b.w;
    reinterpret_cast<float4*>(out)[t] = o;
}

// ---------------------------------------------------------------------------
// Edge scatter: 16 threads per edge, float4 gather, 4x f32 atomic scatter.
// norm = dinv[row] * w * dinv[col];  out[col] += norm * H[row]
// ---------------------------------------------------------------------------
__global__ __launch_bounds__(256)
void gcn_edge_scatter(const float* __restrict__ H,
                      const int* __restrict__ row,
                      const int* __restrict__ col,
                      const float* __restrict__ w,
                      const float* __restrict__ dinv,
                      float* __restrict__ out, int e) {
    int t = blockIdx.x * blockDim.x + threadIdx.x;
    const int ei = t >> 4;
    if (ei >= e) return;
    const int c4 = t & 15;
    const int r = row[ei], c = col[ei];
    const float nrm = dinv[r] * w[ei] * dinv[c];
    float4 v = reinterpret_cast<const float4*>(H + (size_t)r * FDIM)[c4];
    float* dst = out + (size_t)c * FDIM + c4 * 4;
    atomicAdd(dst + 0, nrm * v.x);
    atomicAdd(dst + 1, nrm * v.y);
    atomicAdd(dst + 2, nrm * v.z);
    atomicAdd(dst + 3, nrm * v.w);
}

__global__ void gcn_relu(float* __restrict__ a, int n16) {
    int t = blockIdx.x * blockDim.x + threadIdx.x;
    if (t >= n16) return;
    float4 v = reinterpret_cast<float4*>(a)[t];
    v.x = fmaxf(v.x, 0.f); v.y = fmaxf(v.y, 0.f);
    v.z = fmaxf(v.z, 0.f); v.w = fmaxf(v.w, 0.f);
    reinterpret_cast<float4*>(a)[t] = v;
}

// ---------------------------------------------------------------------------
extern "C" void kernel_launch(void* const* d_in, const int* in_sizes, int n_in,
                              void* d_out, int out_size, void* d_ws, size_t ws_size,
                              hipStream_t stream) {
    const float* x   = (const float*)d_in[0];
    const float* ew  = (const float*)d_in[1];
    const float* W1  = (const float*)d_in[2];
    const float* b1  = (const float*)d_in[3];
    const float* W2  = (const float*)d_in[4];
    const float* b2  = (const float*)d_in[5];
    const int*  eidx = (const int*)d_in[6];

    const int n = in_sizes[0] / FDIM;       // 50000
    const int e = in_sizes[1];              // 800000
    const int* row = eidx;                  // edge_index[0] = source
    const int* col = eidx + e;              // edge_index[1] = target

    float* deg  = (float*)d_ws;
    float* dinv = deg  + n;
    float* hlin = dinv + n;                       // N*64, reused for both layers
    float* agg1 = hlin + (size_t)n * FDIM;        // N*64
    float* outp = (float*)d_out;

    const int B = 256;
    const int gN   = (n + B - 1) / B;
    const int gE   = (e + B - 1) / B;
    const int gN16 = (n * 16 + B - 1) / B;
    const int gE16 = (e * 16 + B - 1) / B;
    const int gGemm = (n + 31) / 32;

    // --- normalization ---
    gcn_deg_init<<<gN, B, 0, stream>>>(deg, n);
    gcn_deg_acc <<<gE, B, 0, stream>>>(col, ew, deg, e);
    gcn_dinv    <<<gN, B, 0, stream>>>(deg, dinv, n);

    // --- layer 1 ---
    gcn_gemm64      <<<gGemm, B, 0, stream>>>(x, W1, hlin, n);
    gcn_init_agg    <<<gN16, B, 0, stream>>>(hlin, dinv, b1, agg1, n);
    gcn_edge_scatter<<<gE16, B, 0, stream>>>(hlin, row, col, ew, dinv, agg1, e);
    gcn_relu        <<<gN16, B, 0, stream>>>(agg1, n * 16);

    // --- layer 2 ---
    gcn_gemm64      <<<gGemm, B, 0, stream>>>(agg1, W2, hlin, n);
    gcn_init_agg    <<<gN16, B, 0, stream>>>(hlin, dinv, b2, outp, n);
    gcn_edge_scatter<<<gE16, B, 0, stream>>>(hlin, row, col, ew, dinv, outp, e);
}
